// ScanAttention_50766513438945
// MI455X (gfx1250) — matrix operand order repaired and checked by reading, compiled, not run on hardware
//
#include <hip/hip_runtime.h>

// ---------------------------------------------------------------------------
// FlashAttention-2 forward for [B=4, S=2048, H=16, D=64], fp32 in/out,
// f16 WMMA (v_wmma_f32_16x16x32_f16) compute on gfx1250 (wave32).
// 8 waves / 128 q-rows per workgroup; 64-key KV tiles staged in LDS as f16.
// ---------------------------------------------------------------------------

typedef __attribute__((ext_vector_type(16))) _Float16 v16h;
typedef __attribute__((ext_vector_type(8)))  _Float16 v8h;
typedef __attribute__((ext_vector_type(4)))  _Float16 v4h;
typedef __attribute__((ext_vector_type(8)))  float    v8f;

static constexpr int Bc = 4;
static constexpr int Sc = 2048;
static constexpr int Hc = 16;
static constexpr int Dc = 64;

static constexpr int NW   = 8;    // waves per block
static constexpr int NT   = NW * 32;   // threads per block (256)
static constexpr int BQ   = NW * 16;   // q rows per workgroup (128)
static constexpr int BK   = 64;   // keys per kv tile
static constexpr int LSTR = 72;   // LDS row stride in halves (pad vs banks)

__device__ __forceinline__ v16h cat8(v8h lo, v8h hi) {
  return __builtin_shufflevector(lo, hi, 0, 1, 2, 3, 4, 5, 6, 7,
                                         8, 9, 10, 11, 12, 13, 14, 15);
}

// B-fragment (K x 16, interleaved K map): halves 0-7 at p, halves 8-15 at p+16.
__device__ __forceinline__ v16h load_b_frag(const _Float16* p) {
  v8h lo = *(const v8h*)(p);
  v8h hi = *(const v8h*)(p + 16);
  return cat8(lo, hi);
}

// A-fragment (16 x K, interleaved K map): halves 0-7 at p, halves 8-15 at p+16.
__device__ __forceinline__ v16h load_a_frag(const _Float16* p) {
  v8h lo = *(const v8h*)(p);
  v8h hi = *(const v8h*)(p + 16);
  return cat8(lo, hi);
}

__global__ __launch_bounds__(NT) void
fa2_fwd_f16wmma(const float* __restrict__ qs, const float* __restrict__ ks,
                const float* __restrict__ vs, const int* __restrict__ vlens,
                float* __restrict__ out) {
  __shared__ __align__(16) _Float16 sK[BK * LSTR];        // K tile  [key][d]
  __shared__ __align__(16) _Float16 sVt[Dc * LSTR];       // V tile  [d][key]
  __shared__ __align__(16) _Float16 sP[NW * 16 * LSTR];   // P tiles [wave][row][key]

  const int tid  = threadIdx.x;
  const int lane = tid & 31;
  const int wave = tid >> 5;
  const int ln   = lane & 15;        // column / half-wave lane
  const int hw   = lane >> 4;        // 0 = lanes 0-15, 1 = lanes 16-31
  const int off  = hw * 8;           // A-layout interleave offset

  const int qt = blockIdx.x;         // q tile
  const int h  = blockIdx.y;
  const int b  = blockIdx.z;
  const int q0 = qt * BQ;

  const int vlen   = vlens[b];
  const int ntiles = (vlen + BK - 1) / BK;

  // ---- Load this wave's Q stripe into A fragments (scaled by 1/sqrt(D)) ----
  const int    qrow_s = q0 + wave * 16 + ln;
  const float* qrow   = qs + (((size_t)(b * Sc + qrow_s) * Hc + h) * Dc);
  v16h aq[2];
#pragma unroll
  for (int c = 0; c < 2; ++c) {
    const float* p = qrow + 32 * c + off;
    v16h a;
#pragma unroll
    for (int i = 0; i < 8; ++i) a[i]     = (_Float16)(p[i] * 0.125f);
#pragma unroll
    for (int i = 0; i < 8; ++i) a[i + 8] = (_Float16)(p[i + 16] * 0.125f);
    aq[c] = a;
  }

  // ---- Per-row online-softmax state (row = j + 8*hw of this wave's stripe) --
  float mrow[8], lrow[8];
  v8f   acc[4];
#pragma unroll
  for (int j = 0; j < 8; ++j) { mrow[j] = -1.0e30f; lrow[j] = 0.0f; }
#pragma unroll
  for (int t = 0; t < 4; ++t) acc[t] = (v8f){};

  _Float16* pw = &sP[wave * 16 * LSTR];

  for (int kt = 0; kt < ntiles; ++kt) {
    const int k0 = kt * BK;

    __syncthreads();  // previous tile's LDS readers done

    // ---- Cooperative stage of K (row-major) and V (transposed) as f16 ----
#pragma unroll
    for (int i = 0; i < 4; ++i) {
      const int g   = tid + NT * i;        // 1024 float4 groups per matrix
      const int key = g >> 4;
      const int d4  = (g & 15) << 2;
      const size_t gbase = ((size_t)(b * Sc + k0 + key) * Hc + h) * Dc + d4;

      const float4 k4 = *(const float4*)(ks + gbase);
      v4h hk = {(_Float16)k4.x, (_Float16)k4.y, (_Float16)k4.z, (_Float16)k4.w};
      *(v4h*)(&sK[key * LSTR + d4]) = hk;

      const float4 v4 = *(const float4*)(vs + gbase);
      sVt[(d4 + 0) * LSTR + key] = (_Float16)v4.x;
      sVt[(d4 + 1) * LSTR + key] = (_Float16)v4.y;
      sVt[(d4 + 2) * LSTR + key] = (_Float16)v4.z;
      sVt[(d4 + 3) * LSTR + key] = (_Float16)v4.w;
    }
    __syncthreads();

    // ---- Prefetch next KV tile into cache while we compute this one ----
    if (kt + 1 < ntiles) {
#pragma unroll
      for (int i = 0; i < 4; ++i) {
        const int g   = tid + NT * i;
        const int key = g >> 4;
        const int d4  = (g & 15) << 2;
        const size_t gnext =
            ((size_t)(b * Sc + k0 + BK + key) * Hc + h) * Dc + d4;
        __builtin_prefetch(ks + gnext, 0, 0);
        __builtin_prefetch(vs + gnext, 0, 0);
      }
    }

    // ---- Scores: S = (Q*scale) @ K^T, 4 key subtiles x 2 d-chunks ----
    v8f sc[4];
#pragma unroll
    for (int t = 0; t < 4; ++t) {
      v8f c = (v8f){};
#pragma unroll
      for (int cc = 0; cc < 2; ++cc) {
        v16h bk = load_b_frag(&sK[(16 * t + ln) * LSTR + 32 * cc + 8 * hw]);
        c = __builtin_amdgcn_wmma_f32_16x16x32_f16(false, aq[cc], false, bk,
                                                   (short)0, c, false, false);
      }
      const int key = k0 + 16 * t + ln;
      if (key >= vlen) {
#pragma unroll
        for (int j = 0; j < 8; ++j) c[j] = -1.0e30f;
      }
      sc[t] = c;
    }

    // ---- Online softmax (row reductions across each 16-lane half) ----
    float alpha[8];
#pragma unroll
    for (int j = 0; j < 8; ++j) {
      float v = fmaxf(fmaxf(sc[0][j], sc[1][j]), fmaxf(sc[2][j], sc[3][j]));
#pragma unroll
      for (int msk = 1; msk <= 8; msk <<= 1)
        v = fmaxf(v, __shfl_xor(v, msk, 32));
      const float mn = fmaxf(mrow[j], v);
      alpha[j] = __expf(mrow[j] - mn);
      mrow[j]  = mn;
    }
#pragma unroll
    for (int t = 0; t < 4; ++t)
#pragma unroll
      for (int j = 0; j < 8; ++j)
        sc[t][j] = __expf(sc[t][j] - mrow[j]);
#pragma unroll
    for (int j = 0; j < 8; ++j) {
      float r = (sc[0][j] + sc[1][j]) + (sc[2][j] + sc[3][j]);
#pragma unroll
      for (int msk = 1; msk <= 8; msk <<= 1)
        r += __shfl_xor(r, msk, 32);
      lrow[j] = lrow[j] * alpha[j] + r;
    }
#pragma unroll
    for (int t = 0; t < 4; ++t)
#pragma unroll
      for (int j = 0; j < 8; ++j)
        acc[t][j] *= alpha[j];

    // ---- P: C-layout -> A-layout via this wave's private LDS region ----
#pragma unroll
    for (int t = 0; t < 4; ++t)
#pragma unroll
      for (int j = 0; j < 8; ++j)
        pw[(j + 8 * hw) * LSTR + 16 * t + ln] = (_Float16)sc[t][j];

    // Same-wave RAW through LDS: DS ops are in-order per wave, so a DS-counter
    // drain + compiler fence is enough (no workgroup barrier needed here).
    asm volatile("s_wait_dscnt 0" ::: "memory");

    // ---- O += P @ V : 2 key-chunks x 4 d-subtiles ----
#pragma unroll
    for (int cc = 0; cc < 2; ++cc) {
      v16h ap = load_a_frag(&pw[ln * LSTR + 32 * cc + off]);
#pragma unroll
      for (int t = 0; t < 4; ++t) {
        v16h bv = load_b_frag(&sVt[(16 * t + ln) * LSTR + 32 * cc + 8 * hw]);
        acc[t] = __builtin_amdgcn_wmma_f32_16x16x32_f16(false, ap, false, bv,
                                                        (short)0, acc[t],
                                                        false, false);
      }
    }
  }

  // ---- Finalize: O / l, write fp32 [B,S,H,D] ----
#pragma unroll
  for (int j = 0; j < 8; ++j) {
    const float inv   = 1.0f / lrow[j];
    const int   row_s = q0 + wave * 16 + j + 8 * hw;
    float* orow = out + (((size_t)(b * Sc + row_s) * Hc + h) * Dc);
#pragma unroll
    for (int t = 0; t < 4; ++t)
      orow[16 * t + ln] = acc[t][j] * inv;
  }
}

extern "C" void kernel_launch(void* const* d_in, const int* in_sizes, int n_in,
                              void* d_out, int out_size, void* d_ws,
                              size_t ws_size, hipStream_t stream) {
  const float* qs    = (const float*)d_in[0];
  const float* ks    = (const float*)d_in[1];
  const float* vs    = (const float*)d_in[2];
  const int*   vlens = (const int*)d_in[3];
  float*       out   = (float*)d_out;

  dim3 grid(Sc / BQ, Hc, Bc);  // 16 x 16 x 4 = 1024 workgroups
  dim3 block(NT);              // 8 wave32s
  fa2_fwd_f16wmma<<<grid, block, 0, stream>>>(qs, ks, vs, vlens, out);
}
